// UpdateConvEqui_35167192220113
// MI455X (gfx1250) — matrix operand order repaired and checked by reading, not compiled, not run on hardware
//
#include <hip/hip_runtime.h>
#include <hip/hip_bf16.h>
#include <math.h>

#define N_NODES 5000
#define N_EDGES 25000

typedef __attribute__((ext_vector_type(16))) _Float16 v16h;
typedef __attribute__((ext_vector_type(8)))  float    v8f;

// Branch-free softplus on fast HW transcendentals:
// softplus(x) = max(x,0) + log(1 + exp(-|x|))  (stable at both tails)
__device__ __forceinline__ float softplus_f(float x) {
  return fmaxf(x, 0.0f) + __logf(1.0f + __expf(-fabsf(x)));
}

__device__ __forceinline__ v8f wmma16(v16h a, v16h b, v8f c) {
  // D = A(16x32 f16) * B(32x16 f16) + C(16x16 f32)
  return __builtin_amdgcn_wmma_f32_16x16x32_f16(false, a, false, b, (short)0, c,
                                                false, false);
}

// Build a 16x32 f16 A-operand (K = kBase..kBase+31) from LDS tile [16][64].
// 16-bit A layout: lane m=lane%16 holds row M=m; VGPR r holds K pairs
// {2r,2r+1}+h*16 (contiguous, matching packed B), h = lane half.
__device__ __forceinline__ v16h load_A16(const _Float16* a, int kBase, int lane) {
  const int hh = (lane >> 4) & 1;
  const int mm = lane & 15;
  v16h A;
#pragma unroll
  for (int r = 0; r < 8; ++r) {
    int kp = kBase + hh * 16 + 2 * r;
    A[2 * r]     = a[mm * 64 + kp];
    A[2 * r + 1] = a[mm * 64 + kp + 1];
  }
  return A;
}

// ---------------------------------------------------------------------------
// Repack a [64 x ncols] f32 weight matrix into lane-major f16 B-operand order.
// Packed halves index: ((q*4+nt)*2+kb)*512 + lane*16 + j
//   holds W[k][n], k = kb*32 + (lane>>4)*16 + j, n = q*64 + nt*16 + (lane&15)
// so each lane's v16h B-operand is one contiguous 32-byte load.
// ---------------------------------------------------------------------------
__global__ void repack_b_kernel(const float* __restrict__ W,
                                _Float16* __restrict__ out, int ncols) {
  int idx = blockIdx.x * blockDim.x + threadIdx.x;
  int total = (ncols >> 6) * 4096;
  if (idx >= total) return;
  int j    = idx & 15;
  int lane = (idx >> 4) & 31;
  int kb   = (idx >> 9) & 1;
  int nt   = (idx >> 10) & 3;
  int q    = idx >> 12;
  int k = kb * 32 + (lane >> 4) * 16 + j;
  int n = q * 64 + nt * 16 + (lane & 15);
  out[idx] = (_Float16)W[k * ncols + n];
}

__global__ void zero_kernel(float* p, int n) {
  int i = blockIdx.x * blockDim.x + threadIdx.x;
  if (i < n) p[i] = 0.0f;
}

__global__ void count_kernel(const int* __restrict__ eidx, float* __restrict__ cnt) {
  int e = blockIdx.x * blockDim.x + threadIdx.x;
  if (e < N_EDGES) atomicAdd(&cnt[eidx[e]], 1.0f);
}

__global__ void sh_kernel(const float* __restrict__ evec, float* __restrict__ shb) {
  int e = blockIdx.x * blockDim.x + threadIdx.x;
  if (e >= N_EDGES) return;
  float x = evec[e * 3 + 0], y = evec[e * 3 + 1], z = evec[e * 3 + 2];
  float r = sqrtf(x * x + y * y + z * z);
  float inv = 1.0f / fmaxf(r, 1e-12f);
  x *= inv; y *= inv; z *= inv;
  const float s3   = 1.7320508075688772f;   // sqrt(3)
  const float s15  = 3.8729833462074170f;   // sqrt(15)
  const float s5h  = 1.1180339887498949f;   // sqrt(5)/2
  const float s15h = 1.9364916731037085f;   // sqrt(15)/2
  float* o = shb + (size_t)e * 8;
  o[0] = s3 * x;  o[1] = s3 * y;  o[2] = s3 * z;
  o[3] = s15 * x * y;
  o[4] = s15 * y * z;
  o[5] = s5h * (3.0f * z * z - 1.0f);
  o[6] = s15 * x * z;
  o[7] = s15h * (x * x - y * y);
}

// h = node_feature @ node_lin_w + b   (one block of 64 threads per node)
__global__ __launch_bounds__(64)
void node_lin_kernel(const float* __restrict__ nf, const float* __restrict__ W,
                     const float* __restrict__ b, float* __restrict__ h) {
  __shared__ float row[64];
  int v = blockIdx.x, t = threadIdx.x;
  row[t] = nf[(size_t)v * 64 + t];
  __syncthreads();
  float acc = b[t];
#pragma unroll 8
  for (int c = 0; c < 64; ++c) acc += row[c] * W[c * 64 + t];
  h[(size_t)v * 64 + t] = acc;
}

// ---------------------------------------------------------------------------
// Layer 1 edge kernel: 16 edges per block, 4 wave32 waves.
//   a1 = softplus(ef@fc1_w1+b1)   (WMMA)
//   for q in 0..79: Wgen tile [16x64] via 2 WMMA, consumed in registers.
//     q<64 : t0[e,n]      += s[e,q]      * (Wgen+b2)
//     q>=64: t1/t2[e,n&7] += s[e,u(q,n)] * (Wgen+b2)   (LDS partials)
//   scatter 0.125*[t0 | t1*sh1 | t2*sh2] into hsum1 with f32 atomics.
// ---------------------------------------------------------------------------
__global__ __launch_bounds__(128)
void edge_layer1_kernel(const int* __restrict__ eidx, const float* __restrict__ ef,
                        const _Float16* __restrict__ pw1, const float* __restrict__ b1,
                        const _Float16* __restrict__ pw2, const float* __restrict__ b2,
                        const float* __restrict__ hbuf, const float* __restrict__ shb,
                        float* __restrict__ hsum1) {
  __shared__ _Float16 lds_ef[16 * 64];
  __shared__ _Float16 lds_a1[16 * 64];
  __shared__ float    lds_s[16 * 64];
  __shared__ float    lds_t1[16 * 8];
  __shared__ float    lds_t2[16 * 8];
  __shared__ int      lds_src[16];

  const int tid  = threadIdx.x;
  const int lane = tid & 31, wv = tid >> 5;
  const int mm   = lane & 15, hh = lane >> 4;
  const int e0   = blockIdx.x * 16;

  for (int i = tid; i < 16 * 64; i += 128) {
    int e = e0 + (i >> 6), c = i & 63;
    float fe = 0.0f, sv = 0.0f;
    if (e < N_EDGES) {
      fe = ef[(size_t)e * 64 + c];
      int dst = eidx[N_EDGES + e];
      sv = hbuf[(size_t)dst * 64 + c];
    }
    lds_ef[i] = (_Float16)fe;
    lds_s[i]  = sv;
  }
  if (tid < 16) lds_src[tid] = (e0 + tid < N_EDGES) ? eidx[e0 + tid] : -1;
  lds_t1[tid] = 0.0f;
  lds_t2[tid] = 0.0f;
  __syncthreads();

  const int n = wv * 16 + mm;

  // GEMM1: a1 = softplus(ef @ fc1_w1 + b1), f16 result staged in LDS
  {
    v16h aE0 = load_A16(lds_ef, 0, lane);
    v16h aE1 = load_A16(lds_ef, 32, lane);
    const v16h* bp = (const v16h*)pw1;
    v8f c = {};
    c = wmma16(aE0, bp[wv * 64 + lane], c);
    c = wmma16(aE1, bp[wv * 64 + 32 + lane], c);
    float bb = b1[n];
#pragma unroll
    for (int r = 0; r < 8; ++r)
      lds_a1[(r + 8 * hh) * 64 + n] = (_Float16)softplus_f(c[r] + bb);
  }
  __syncthreads();

  v16h A0 = load_A16(lds_a1, 0, lane);
  v16h A1 = load_A16(lds_a1, 32, lane);
  const v16h* w2p = (const v16h*)pw2;

  // q = 0..63: t0 path (output col fixed at n across q)
  v8f acc = {};
#pragma unroll 2
  for (int q = 0; q < 64; ++q) {
    const v16h* bq = w2p + (size_t)(q * 4 + wv) * 64;
    __builtin_prefetch((const void*)(w2p + (size_t)((q + 2) * 4 + wv) * 64), 0, 3);
    v8f d = {};
    d = wmma16(A0, bq[lane], d);
    d = wmma16(A1, bq[32 + lane], d);
    float bc = b2[q * 64 + n];
#pragma unroll
    for (int r = 0; r < 8; ++r)
      acc[r] += lds_s[(r + 8 * hh) * 64 + q] * (d[r] + bc);
  }

  // q = 64..79: t1 (q<72) and t2 (q>=72) 8-wide sub-blocks -> LDS partials
#pragma unroll 2
  for (int q = 64; q < 80; ++q) {
    const v16h* bq = w2p + (size_t)(q * 4 + wv) * 64;
    v8f d = {};
    d = wmma16(A0, bq[lane], d);
    d = wmma16(A1, bq[32 + lane], d);
    float bc = b2[q * 64 + n];
    int u = ((q - 64) & 7) * 8 + (n >> 3);
    int vv = n & 7;
    float* dest = (q < 72) ? lds_t1 : lds_t2;
#pragma unroll
    for (int r = 0; r < 8; ++r) {
      int M = r + 8 * hh;
      atomicAdd(&dest[M * 8 + vv], lds_s[M * 64 + u] * (d[r] + bc));
    }
  }

  // scatter t0 (tp1 cols 0..63)
#pragma unroll
  for (int r = 0; r < 8; ++r) {
    int M = r + 8 * hh;
    int src = lds_src[M];
    if (src >= 0) atomicAdd(&hsum1[(size_t)src * 128 + n], 0.125f * acc[r]);
  }
  __syncthreads();

  // scatter t1*sh1 (cols 64+v*3+m) and t2*sh2 (cols 88+v*5+m)
  {
    int e = tid >> 3, vv = tid & 7;
    int src = lds_src[e];
    if (src >= 0) {
      const float* shE = shb + (size_t)(e0 + e) * 8;
      float t1v = 0.125f * lds_t1[e * 8 + vv];
      float t2v = 0.125f * lds_t2[e * 8 + vv];
#pragma unroll
      for (int m2 = 0; m2 < 3; ++m2)
        atomicAdd(&hsum1[(size_t)src * 128 + 64 + vv * 3 + m2], t1v * shE[m2]);
#pragma unroll
      for (int m2 = 0; m2 < 5; ++m2)
        atomicAdd(&hsum1[(size_t)src * 128 + 88 + vv * 5 + m2], t2v * shE[3 + m2]);
    }
  }
}

__global__ void finalize_h1_kernel(const float* __restrict__ hsum1,
                                   const float* __restrict__ cnt,
                                   const float* __restrict__ hbuf,
                                   float* __restrict__ h1) {
  int i = blockIdx.x * blockDim.x + threadIdx.x;
  if (i >= N_NODES * 128) return;
  int v = i >> 7, c = i & 127;
  float x = hsum1[i] / fmaxf(cnt[v], 1.0f);
  if (c < 64) x += hbuf[(size_t)v * 64 + c];
  h1[i] = x;
}

// ---------------------------------------------------------------------------
// Layer 2 edge kernel. Uniform structure: y[e,n] = sum_q coef[e,q]*(Wgen+b2),
// coef = [ s0 (64) | (v1·sh1)/sqrt3 (8) | (v2·sh2)/sqrt5 (8) ], q = 0..79.
// ---------------------------------------------------------------------------
__global__ __launch_bounds__(128)
void edge_layer2_kernel(const int* __restrict__ eidx, const float* __restrict__ ef,
                        const _Float16* __restrict__ pw1, const float* __restrict__ b1,
                        const _Float16* __restrict__ pw2, const float* __restrict__ b2,
                        const float* __restrict__ h1, const float* __restrict__ shb,
                        float* __restrict__ h2sum) {
  __shared__ _Float16 lds_ef[16 * 64];
  __shared__ _Float16 lds_a2[16 * 64];
  __shared__ float    lds_coef[16 * 80];
  __shared__ int      lds_src[16];

  const int tid  = threadIdx.x;
  const int lane = tid & 31, wv = tid >> 5;
  const int mm   = lane & 15, hh = lane >> 4;
  const int e0   = blockIdx.x * 16;

  for (int i = tid; i < 16 * 64; i += 128) {
    int e = e0 + (i >> 6), c = i & 63;
    float fe = 0.0f, cf = 0.0f;
    if (e < N_EDGES) {
      fe = ef[(size_t)e * 64 + c];
      int dst = eidx[N_EDGES + e];
      cf = h1[(size_t)dst * 128 + c];          // s0 part
    }
    lds_ef[i] = (_Float16)fe;
    lds_coef[(i >> 6) * 80 + c] = cf;
  }
  {
    int e = tid >> 3, p = tid & 7;
    float c1 = 0.0f, c2 = 0.0f;
    int eg = e0 + e;
    if (eg < N_EDGES) {
      int dst = eidx[N_EDGES + eg];
      const float* shE = shb + (size_t)eg * 8;
#pragma unroll
      for (int m2 = 0; m2 < 3; ++m2)
        c1 += h1[(size_t)dst * 128 + 64 + p * 3 + m2] * shE[m2];
#pragma unroll
      for (int m2 = 0; m2 < 5; ++m2)
        c2 += h1[(size_t)dst * 128 + 88 + p * 5 + m2] * shE[3 + m2];
    }
    lds_coef[e * 80 + 64 + p] = c1 * 0.5773502691896258f;  // 1/sqrt(3)
    lds_coef[e * 80 + 72 + p] = c2 * 0.4472135954999579f;  // 1/sqrt(5)
  }
  if (tid < 16) lds_src[tid] = (e0 + tid < N_EDGES) ? eidx[e0 + tid] : -1;
  __syncthreads();

  const int n = wv * 16 + mm;

  // a2 = softplus(ef @ fc2_w1 + b1)
  {
    v16h aE0 = load_A16(lds_ef, 0, lane);
    v16h aE1 = load_A16(lds_ef, 32, lane);
    const v16h* bp = (const v16h*)pw1;
    v8f c = {};
    c = wmma16(aE0, bp[wv * 64 + lane], c);
    c = wmma16(aE1, bp[wv * 64 + 32 + lane], c);
    float bb = b1[n];
#pragma unroll
    for (int r = 0; r < 8; ++r)
      lds_a2[(r + 8 * hh) * 64 + n] = (_Float16)softplus_f(c[r] + bb);
  }
  __syncthreads();

  v16h A0 = load_A16(lds_a2, 0, lane);
  v16h A1 = load_A16(lds_a2, 32, lane);
  const v16h* w2p = (const v16h*)pw2;

  v8f acc = {};
#pragma unroll 2
  for (int q = 0; q < 80; ++q) {
    const v16h* bq = w2p + (size_t)(q * 4 + wv) * 64;
    if (q + 2 < 80)
      __builtin_prefetch((const void*)(w2p + (size_t)((q + 2) * 4 + wv) * 64), 0, 3);
    v8f d = {};
    d = wmma16(A0, bq[lane], d);
    d = wmma16(A1, bq[32 + lane], d);
    float bc = b2[q * 64 + n];
#pragma unroll
    for (int r = 0; r < 8; ++r)
      acc[r] += lds_coef[(r + 8 * hh) * 80 + q] * (d[r] + bc);
  }

  const float inv_s80 = 0.11180339887498949f;  // 1/sqrt(80)
#pragma unroll
  for (int r = 0; r < 8; ++r) {
    int M = r + 8 * hh;
    int src = lds_src[M];
    if (src >= 0) atomicAdd(&h2sum[(size_t)src * 64 + n], inv_s80 * acc[r]);
  }
}

__global__ __launch_bounds__(256)
void bn_stats_kernel(const float* __restrict__ h2sum, const float* __restrict__ cnt,
                     float* __restrict__ mu, float* __restrict__ var) {
  __shared__ float ss[256], sq[256];
  int c = blockIdx.x, t = threadIdx.x;
  float s = 0.0f, q = 0.0f;
  for (int v = t; v < N_NODES; v += 256) {
    float x = h2sum[(size_t)v * 64 + c] / fmaxf(cnt[v], 1.0f);
    s += x; q += x * x;
  }
  ss[t] = s; sq[t] = q;
  __syncthreads();
  for (int o = 128; o > 0; o >>= 1) {
    if (t < o) { ss[t] += ss[t + o]; sq[t] += sq[t + o]; }
    __syncthreads();
  }
  if (t == 0) {
    float m = ss[0] / (float)N_NODES;
    mu[c]  = m;
    var[c] = sq[0] / (float)N_NODES - m * m;
  }
}

// out = softplus(softplus(bn(h2)) @ lin2_w + b) + node_feature
__global__ __launch_bounds__(64)
void out_kernel(const float* __restrict__ h2sum, const float* __restrict__ cnt,
                const float* __restrict__ mu, const float* __restrict__ var,
                const float* __restrict__ gamma, const float* __restrict__ beta,
                const float* __restrict__ lw, const float* __restrict__ lb,
                const float* __restrict__ nf, float* __restrict__ out) {
  __shared__ float sp[64];
  int v = blockIdx.x, t = threadIdx.x;
  float x  = h2sum[(size_t)v * 64 + t] / fmaxf(cnt[v], 1.0f);
  float hn = gamma[t] * (x - mu[t]) * rsqrtf(var[t] + 1e-5f) + beta[t];
  sp[t] = softplus_f(hn);
  __syncthreads();
  float acc = lb[t];
#pragma unroll 8
  for (int c = 0; c < 64; ++c) acc += sp[c] * lw[c * 64 + t];
  out[(size_t)v * 64 + t] = softplus_f(acc) + nf[(size_t)v * 64 + t];
}

extern "C" void kernel_launch(void* const* d_in, const int* in_sizes, int n_in,
                              void* d_out, int out_size, void* d_ws, size_t ws_size,
                              hipStream_t stream) {
  (void)in_sizes; (void)n_in; (void)out_size; (void)ws_size;
  const float* nf   = (const float*)d_in[0];
  const int*   eidx = (const int*)d_in[1];
  const float* ef   = (const float*)d_in[2];
  const float* evec = (const float*)d_in[3];
  const float* nlw  = (const float*)d_in[4];
  const float* nlb  = (const float*)d_in[5];
  const float* f1w1 = (const float*)d_in[6];
  const float* f1b1 = (const float*)d_in[7];
  const float* f1w2 = (const float*)d_in[8];
  const float* f1b2 = (const float*)d_in[9];
  const float* f2w1 = (const float*)d_in[10];
  const float* f2b1 = (const float*)d_in[11];
  const float* f2w2 = (const float*)d_in[12];
  const float* f2b2 = (const float*)d_in[13];
  const float* gam  = (const float*)d_in[14];
  const float* bet  = (const float*)d_in[15];
  const float* l2w  = (const float*)d_in[16];
  const float* l2b  = (const float*)d_in[17];
  float* out = (float*)d_out;

  char* wsp = (char*)d_ws;
  size_t off = 0;
  auto alloc = [&](size_t bytes) {
    void* p = wsp + off;
    off = (off + bytes + 255) & ~(size_t)255;
    return p;
  };
  _Float16* pw1a = (_Float16*)alloc(4096 * 2);            // fc1_w1 packed
  _Float16* pw2a = (_Float16*)alloc(4096 * 2);            // fc2_w1 packed
  _Float16* pw1b = (_Float16*)alloc((size_t)327680 * 2);  // fc1_w2 packed
  _Float16* pw2b = (_Float16*)alloc((size_t)327680 * 2);  // fc2_w2 packed
  float* hbuf  = (float*)alloc((size_t)N_NODES * 64 * 4);
  float* hsum1 = (float*)alloc((size_t)N_NODES * 128 * 4);
  float* h1buf = (float*)alloc((size_t)N_NODES * 128 * 4);
  float* h2sum = (float*)alloc((size_t)N_NODES * 64 * 4);
  float* cnt   = (float*)alloc((size_t)N_NODES * 4);
  float* shb   = (float*)alloc((size_t)N_EDGES * 8 * 4);
  float* mu    = (float*)alloc(64 * 4);
  float* var   = (float*)alloc(64 * 4);

  zero_kernel<<<(N_NODES * 128 + 255) / 256, 256, 0, stream>>>(hsum1, N_NODES * 128);
  zero_kernel<<<(N_NODES * 64 + 255) / 256, 256, 0, stream>>>(h2sum, N_NODES * 64);
  zero_kernel<<<(N_NODES + 255) / 256, 256, 0, stream>>>(cnt, N_NODES);

  repack_b_kernel<<<(4096 + 255) / 256, 256, 0, stream>>>(f1w1, pw1a, 64);
  repack_b_kernel<<<(4096 + 255) / 256, 256, 0, stream>>>(f2w1, pw2a, 64);
  repack_b_kernel<<<(327680 + 255) / 256, 256, 0, stream>>>(f1w2, pw1b, 5120);
  repack_b_kernel<<<(327680 + 255) / 256, 256, 0, stream>>>(f2w2, pw2b, 5120);

  sh_kernel<<<(N_EDGES + 255) / 256, 256, 0, stream>>>(evec, shb);
  count_kernel<<<(N_EDGES + 255) / 256, 256, 0, stream>>>(eidx, cnt);
  node_lin_kernel<<<N_NODES, 64, 0, stream>>>(nf, nlw, nlb, hbuf);

  const int nTiles = (N_EDGES + 15) / 16;
  edge_layer1_kernel<<<nTiles, 128, 0, stream>>>(eidx, ef, pw1a, f1b1, pw1b, f1b2,
                                                 hbuf, shb, hsum1);
  finalize_h1_kernel<<<(N_NODES * 128 + 255) / 256, 256, 0, stream>>>(hsum1, cnt,
                                                                      hbuf, h1buf);
  edge_layer2_kernel<<<nTiles, 128, 0, stream>>>(eidx, ef, pw2a, f2b1, pw2b, f2b2,
                                                 h1buf, shb, h2sum);
  bn_stats_kernel<<<64, 256, 0, stream>>>(h2sum, cnt, mu, var);
  out_kernel<<<N_NODES, 64, 0, stream>>>(h2sum, cnt, mu, var, gam, bet, l2w, l2b,
                                         nf, out);
}
